// SharedBottom_41575283425675
// MI455X (gfx1250) — matrix operand order repaired and checked by reading, not compiled, not run on hardware
//
#include <hip/hip_runtime.h>
#include <hip/hip_bf16.h>

typedef __attribute__((ext_vector_type(16))) _Float16 v16h;
typedef __attribute__((ext_vector_type(8)))  float    v8f;

// Problem constants (from reference)
constexpr int cB   = 16384;
constexpr int cFIN = 640;
constexpr int cH1  = 1024;
constexpr int cH2  = 512;
constexpr int cT1  = 256;
constexpr int cT2  = 128;
constexpr int cD   = 8;

// Tiling
#define BLK_M 64
#define BLK_N 128
#define KB    32
#define NTILES (BLK_N / 16)   // 8 16-wide B tiles per block

// C = relu(A[M,K] * Bw[K,N] + bias[N]), output f16.
// A is either fp32 (stage 1: x) or f16 (workspace intermediates).
// Bw/bias are the original fp32 weights (L2-resident); B tile is staged into
// LDS in FRAGMENT-READY layout: Bs[tile][lane][16 halves contiguous], where
// lane = (n%16) + 16*(k/16) and half h holds k = 16*(k/16) + h. Compute-side
// fragment loads are then a single contiguous 32B read per lane.
template <typename AT>
__global__ __launch_bounds__(256)
void wmma_gemm_bias_relu(const AT* __restrict__ A,
                         const float* __restrict__ Bw,
                         const float* __restrict__ bias,
                         _Float16* __restrict__ C,
                         int M, int N, int K)
{
    __shared__ __align__(32) _Float16 As[BLK_M][KB + 8];          // 64x40 halves
    __shared__ __align__(32) _Float16 Bs[NTILES * 32 * 16];       // 4096 halves, frag-ready

    const int tid  = threadIdx.x;
    const int wave = tid >> 5;      // 0..7
    const int lane = tid & 31;
    const int wm   = wave & 3;      // 4 waves tile M (4*16 = 64)
    const int wn   = wave >> 2;     // 2 waves tile N (2*64 = 128)
    const int m0   = blockIdx.y * BLK_M;
    const int n0   = blockIdx.x * BLK_N;

    const int l16 = lane & 15;
    const int hi  = lane >> 4;      // lane half select (k-group)

    v8f acc0 = {}, acc1 = {}, acc2 = {}, acc3 = {};

    // Per-thread staging assignments (loop-invariant)
    const int a_row = tid >> 2;            // 0..63
    const int a_col = (tid & 3) * 8;       // 0,8,16,24
    const int b_n   = tid & (BLK_N - 1);   // 0..127 (consecutive threads -> consecutive n: coalesced)
    const int b_kh  = tid >> 7;            // 0..1  (which 16-wide k group)
    _Float16* b_dst = &Bs[(((b_n >> 4) * 32) + (b_n & 15) + 16 * b_kh) * 16];

    for (int k0 = 0; k0 < K; k0 += KB) {
        // --- stage A tile: 64x32 halves, 8 contiguous per thread ---
        {
            const AT* src = A + (long)(m0 + a_row) * K + k0 + a_col;
            #pragma unroll
            for (int i = 0; i < 8; ++i)
                As[a_row][a_col + i] = (_Float16)src[i];
        }
        // --- stage B tile: each thread gathers a 16-deep K column for col b_n ---
        {
            const float* src = Bw + (long)(k0 + b_kh * 16) * N + n0 + b_n;
            v16h breg;
            #pragma unroll
            for (int j = 0; j < 16; ++j)
                breg[j] = (_Float16)src[(long)j * N];
            *(v16h*)b_dst = breg;          // 32B contiguous store
        }
        __syncthreads();

        // --- A fragment (16x32 MxK per ISA):
        // lanes 0-15: K 0-15 ; lanes 16-31: K 16-31
        v16h afrag;
        {
            const int row = wm * 16 + l16;
            #pragma unroll
            for (int h = 0; h < 8; ++h)  afrag[h] = As[row][16 * hi + h];
            #pragma unroll
            for (int h = 8; h < 16; ++h) afrag[h] = As[row][16 * hi + h];
        }
        // --- 4 fragment-ready B loads + WMMA ---
        #pragma unroll
        for (int t = 0; t < 4; ++t) {
            v16h bfrag = *(const v16h*)&Bs[(((wn * 4 + t) * 32) + lane) * 16];
            v8f* accp = (t == 0) ? &acc0 : (t == 1) ? &acc1 : (t == 2) ? &acc2 : &acc3;
            *accp = __builtin_amdgcn_wmma_f32_16x16x32_f16(
                        false, afrag, false, bfrag, (short)0, *accp, false, false);
        }
        __syncthreads();
    }

    // --- epilogue: bias + relu, store f16 ---
    v8f accs[4] = {acc0, acc1, acc2, acc3};
    #pragma unroll
    for (int t = 0; t < 4; ++t) {
        const int n  = n0 + wn * 64 + t * 16 + l16;
        const float bv = bias[n];
        #pragma unroll
        for (int r = 0; r < 8; ++r) {
            const int m = m0 + wm * 16 + r + 8 * hi;
            float v = accs[t][r] + bv;
            v = v > 0.0f ? v : 0.0f;
            C[(long)m * N + n] = (_Float16)v;
        }
    }
}

// out[b] = sigmoid( dot(T2[d,b,:], TWo[d,:,0]) + Tbo[d] ), d = domain_id[b]
__global__ __launch_bounds__(256)
void tower_out_kernel(const _Float16* __restrict__ T2buf,  // [D,B,128] f16
                      const float* __restrict__ TWo,       // [D,128,1]
                      const float* __restrict__ Tbo,       // [D,1]
                      const int* __restrict__ domain_id,   // [B]
                      float* __restrict__ out, int Bn)
{
    const int b = blockIdx.x * blockDim.x + threadIdx.x;
    if (b >= Bn) return;
    const int d = domain_id[b];
    const _Float16* t = T2buf + ((long)d * Bn + b) * cT2;
    const float*    w = TWo + (long)d * cT2;
    float s = 0.0f;
    #pragma unroll 8
    for (int k = 0; k < cT2; ++k)
        s += (float)t[k] * w[k];
    s += Tbo[d];
    out[b] = 1.0f / (1.0f + expf(-s));
}

extern "C" void kernel_launch(void* const* d_in, const int* in_sizes, int n_in,
                              void* d_out, int out_size, void* d_ws, size_t ws_size,
                              hipStream_t stream)
{
    const float* x    = (const float*)d_in[0];
    const int*   dom  = (const int*)  d_in[1];
    const float* W1   = (const float*)d_in[2];
    const float* b1   = (const float*)d_in[3];
    const float* W2   = (const float*)d_in[4];
    const float* b2   = (const float*)d_in[5];
    const float* TW1  = (const float*)d_in[6];
    const float* Tb1  = (const float*)d_in[7];
    const float* TW2  = (const float*)d_in[8];
    const float* Tb2  = (const float*)d_in[9];
    const float* TWo  = (const float*)d_in[10];
    const float* Tbo  = (const float*)d_in[11];
    float* out = (float*)d_out;

    // Workspace layout (f16):
    //   buf0: B*H1 halves (32MB)  — holds H1 for stage 2, then reused as T2[D,B,128]
    //         (D*B*T2 == B*H1 == 16,777,216 halves, exact fit)
    //   buf1: B*H2 halves (16MB)  — H2
    //   buf2: B*T1 halves (8MB)   — per-domain T1 scratch
    char* ws = (char*)d_ws;
    _Float16* buf0 = (_Float16*)ws;
    _Float16* buf1 = (_Float16*)(ws + (size_t)cB * cH1 * sizeof(_Float16));
    _Float16* buf2 = (_Float16*)(ws + (size_t)cB * cH1 * sizeof(_Float16)
                                    + (size_t)cB * cH2 * sizeof(_Float16));

    const dim3 blk(256);

    // Stage 1: H1 = relu(x @ W1 + b1)    [16384,640]x[640,1024]
    wmma_gemm_bias_relu<float><<<dim3(cH1 / BLK_N, cB / BLK_M), blk, 0, stream>>>(
        x, W1, b1, buf0, cB, cH1, cFIN);

    // Stage 2: H2 = relu(H1 @ W2 + b2)   [16384,1024]x[1024,512]
    wmma_gemm_bias_relu<_Float16><<<dim3(cH2 / BLK_N, cB / BLK_M), blk, 0, stream>>>(
        buf0, W2, b2, buf1, cB, cH2, cH1);

    // Per-domain towers (H1 in buf0 is dead now; buf0 becomes T2[D,B,128])
    for (int d = 0; d < cD; ++d) {
        // T1 = relu(H2 @ TW1[d] + Tb1[d])   [16384,512]x[512,256]
        wmma_gemm_bias_relu<_Float16><<<dim3(cT1 / BLK_N, cB / BLK_M), blk, 0, stream>>>(
            buf1, TW1 + (long)d * cH2 * cT1, Tb1 + (long)d * cT1, buf2, cB, cT1, cH2);
        // T2[d] = relu(T1 @ TW2[d] + Tb2[d]) [16384,256]x[256,128]
        wmma_gemm_bias_relu<_Float16><<<dim3(cT2 / BLK_N, cB / BLK_M), blk, 0, stream>>>(
            buf2, TW2 + (long)d * cT1 * cT2, Tb2 + (long)d * cT2,
            buf0 + (size_t)d * cB * cT2, cB, cT2, cT1);
    }

    // Final: per-sample GEMV over selected domain + sigmoid
    tower_out_kernel<<<(cB + 255) / 256, 256, 0, stream>>>(
        buf0, TWo, Tbo, dom, out, cB);
}
